// CompositionalAttentionBase_25451976196695
// MI455X (gfx1250) — matrix operand order repaired and checked by reading, not compiled, not run on hardware
//
#include <hip/hip_runtime.h>

typedef __attribute__((ext_vector_type(16))) _Float16 v16h;
typedef __attribute__((ext_vector_type(8)))  _Float16 v8h;
typedef __attribute__((ext_vector_type(8)))  float    v8f;

#define DEV static __device__ __forceinline__

DEV v16h pack16(v8h lo, v8h hi) {
  v16h r;
#pragma unroll
  for (int i = 0; i < 8; ++i) { r[i] = lo[i]; r[i + 8] = hi[i]; }
  return r;
}

// A fragment (16x32 f16) from a row-major tile (pitch in halfs).
// lane: m = l&15; elements 0..7 = A[m][8h..8h+7], 8..15 = A[m][16+8h..+7], h = l>>4.
DEV v16h frag_a(const _Float16* tile, int pitch) {
  const int lane = threadIdx.x & 31;
  const int m = lane & 15, h8 = (lane >> 4) * 8;
  v8h lo = *(const v8h*)(tile + m * pitch + h8);
  v8h hi = *(const v8h*)(tile + m * pitch + 16 + h8);
  return pack16(lo, hi);
}

// B fragment (32x16 f16) from transposed storage tileT[n][k] (pitch in halfs).
// lane: n = l&15; elements 0..7 = B[8h..8h+7][n], 8..15 = B[16+8h..+7][n], h = l>>4.
DEV v16h frag_b_trans(const _Float16* tileT, int pitch) {
  const int lane = threadIdx.x & 31;
  const int n = lane & 15, k0 = (lane >> 4) * 8;
  v8h lo = *(const v8h*)(tileT + n * pitch + k0);
  v8h hi = *(const v8h*)(tileT + n * pitch + k0 + 16);
  return pack16(lo, hi);
}

DEV v8f wmma_f16(v16h a, v16h b, v8f c) {
  return __builtin_amdgcn_wmma_f32_16x16x32_f16(false, a, false, b, (short)0, c,
                                                false, false);
}

// CDNA5 async copy: global memory -> LDS, 16B per lane, tracked by ASYNCcnt.
// LDS offset = low 32 bits of the generic pointer (LDS aperture truncation).
DEV void async_b128(const void* lds_dst, const void* gsrc) {
  asm volatile("global_load_async_to_lds_b128 %0, %1, off"
               :
               : "v"((unsigned)(uintptr_t)lds_dst),
                 "v"((unsigned long long)(uintptr_t)gsrc)
               : "memory");
}
DEV void wait_async0() { asm volatile("s_wait_asynccnt 0x0" ::: "memory"); }

// ---------------------------------------------------------------------------
// Elementwise f32 -> f16 conversion.
// ---------------------------------------------------------------------------
__global__ void cvt_f16(const float* __restrict__ in, _Float16* __restrict__ out, int n) {
  const int i = blockIdx.x * 256 + threadIdx.x;
  if (i < n) out[i] = (_Float16)in[i];
}

// ---------------------------------------------------------------------------
// Tiled WMMA GEMM:  out = (A[MxK] @ B[KxN] + bias) * alpha
// BLOCK 128x128, K-step 32, 256 threads = 8 waves (4x2), wave tile 32x64.
// A staged via async-to-LDS (row major); B staged transposed (Bt[n][k]) so
// B fragments are two contiguous ds_load_b128.
// Dynamic LDS: As 128x40 halfs @0, Bt 128x40 halfs @10240. Total 20480 B.
// ---------------------------------------------------------------------------
template <bool OUTF32>
__global__ void __launch_bounds__(256)
gemm16(const _Float16* __restrict__ A, const _Float16* __restrict__ B,
       const float* __restrict__ bias, float alpha,
       _Float16* __restrict__ outH, float* __restrict__ outF,
       int M, int N, int K) {
  extern __shared__ char smem[];
  _Float16* As = (_Float16*)smem;            // 128 x (32+8)
  _Float16* Bt = (_Float16*)(smem + 10240);  // 128 n-rows x (32+8)
  const int tid = threadIdx.x, lane = tid & 31, w = tid >> 5;
  const int wm = w >> 1, wn = w & 1;
  const int m0 = blockIdx.y * 128, n0 = blockIdx.x * 128;
  v8f acc[2][4] = {};

  for (int k0 = 0; k0 < K; k0 += 32) {
    __syncthreads();
    { // A tile 128x32: async global->LDS, row major
      const int row = tid >> 1, seg = (tid & 1) * 16;
      const _Float16* s = A + (size_t)(m0 + row) * K + k0 + seg;
      async_b128(&As[row * 40 + seg], s);
      async_b128(&As[row * 40 + seg + 8], s + 8);
    }
    { // B tile 32x128: load row-major, scatter transposed into Bt[n][k]
      const int kk = tid >> 3, nb = (tid & 7) * 16;
      const _Float16* s = B + (size_t)(k0 + kk) * N + n0 + nb;
      v8h b0 = *(const v8h*)s, b1 = *(const v8h*)(s + 8);
#pragma unroll
      for (int j = 0; j < 8; ++j) {
        Bt[(nb + j) * 40 + kk]     = b0[j];
        Bt[(nb + 8 + j) * 40 + kk] = b1[j];
      }
    }
    wait_async0();
    __syncthreads();
    const v16h a0 = frag_a(&As[(wm * 32) * 40], 40);
    const v16h a1 = frag_a(&As[(wm * 32 + 16) * 40], 40);
#pragma unroll
    for (int nf = 0; nf < 4; ++nf) {
      const v16h b = frag_b_trans(Bt + (wn * 64 + nf * 16) * 40, 40);
      acc[0][nf] = wmma_f16(a0, b, acc[0][nf]);
      acc[1][nf] = wmma_f16(a1, b, acc[1][nf]);
    }
  }

  const int half = lane >> 4, nl = lane & 15;
#pragma unroll
  for (int mf = 0; mf < 2; ++mf)
#pragma unroll
    for (int nf = 0; nf < 4; ++nf) {
      const int col = n0 + wn * 64 + nf * 16 + nl;
      const float bv = bias ? bias[col] : 0.0f;
#pragma unroll
      for (int i = 0; i < 8; ++i) {
        const int row = m0 + wm * 32 + mf * 16 + 8 * half + i;
        const float v = (acc[mf][nf][i] + bv) * alpha;
        if (OUTF32) outF[(size_t)row * N + col] = v;
        else        outH[(size_t)row * N + col] = (_Float16)v;
      }
    }
}

// ---------------------------------------------------------------------------
// Fused stage-1 flash attention (all 8 rules) + stage-2 composition softmax.
// Grid: (T/16, B*H). Block: 256 threads = 8 waves; wave w owns rule r = w.
// Q/K tiles staged via async-to-LDS; V tiles staged transposed (Vt[p][k]).
// Wave 0 computes S^T = K.Q^T + online softmax, broadcasts probs/scales in LDS.
// Epilogue computes k_v = out@Wkv, comp softmax over rules, weighted sum.
// ---------------------------------------------------------------------------
__global__ void __launch_bounds__(256)
attn_fused(const _Float16* __restrict__ qh, const _Float16* __restrict__ kh,
           const _Float16* __restrict__ vh, const _Float16* __restrict__ qvh,
           const float* __restrict__ Wkv, const float* __restrict__ bkv,
           _Float16* __restrict__ out2h) {
  extern __shared__ char smem[];
  _Float16* Qs = (_Float16*)smem;          // 16 x 136
  _Float16* Ks = Qs + 16 * 136;            // 32 x 136
  _Float16* Ps = Ks + 32 * 136;            // 16 x 40 probabilities (f16)
  float* fb     = (float*)(Ps + 16 * 40);
  float* scaleS = fb;         // 16  per-q online-softmax rescale
  float* lsumS  = fb + 16;    // 16  per-q softmax denominators
  float* compS  = fb + 32;    // 128 composition logits [q][r]
  float* wS     = fb + 160;   // 128 composition weights [q][r]
  float* bkvS   = fb + 288;   // 32
  float* qvS    = fb + 320;   // 512 q_v [q][32]
  char* dynb    = smem + 17664;
  _Float16* Vs  = (_Float16*)dynb;             // 8 rules x (128 x 40) Vt[p][k]
  float* out_lds = (float*)dynb;               // 16 x 8 x 128 f32 (epilogue reuse)
  float* WkvS    = out_lds + 16 * 8 * 128;     // 128 x 32 f32     (epilogue)

  const int tid = threadIdx.x, lane = tid & 31, w = tid >> 5;
  const int half = lane >> 4, nl = lane & 15;
  const int bh = blockIdx.y, b = bh >> 3, h = bh & 7;
  const int q0 = blockIdx.x * 16;

  const _Float16* qb  = qh + ((size_t)b * 1024 + q0) * 1024 + h * 128;
  const _Float16* kbp = kh + (size_t)b * 1024 * 1024 + h * 128;
  const _Float16* vbp = vh + (size_t)b * 1024 * 8192 + (h * 8 + w) * 128;

  { // stage the 16x128 Q tile once (async)
    const int row = tid >> 4, seg = (tid & 15) * 8;
    async_b128(&Qs[row * 136 + seg], qb + (size_t)row * 1024 + seg);
  }

  v8f acc[8] = {};                 // rule r=w, 8 chunks of 16 over P=128
  float rowm = -3.0e38f, lsum = 0.0f;

  for (int kt = 0; kt < 1024; kt += 32) {
    __syncthreads();
    { // K tile 32x128 row-major: async global->LDS
      const int row = tid >> 3, seg = (tid & 7) * 16;
      const _Float16* s = kbp + (size_t)(kt + row) * 1024 + seg;
      async_b128(&Ks[row * 136 + seg], s);
      async_b128(&Ks[row * 136 + seg + 8], s + 8);
    }
    { // V tile for this wave's rule: stage transposed Vt[p][k]
      _Float16* Vt = Vs + w * (128 * 40);
#pragma unroll
      for (int it = 0; it < 16; ++it) {
        const int chunk = it * 32 + lane;
        const int row = chunk >> 4, seg = (chunk & 15) * 8;
        v8h v = *(const v8h*)(vbp + (size_t)(kt + row) * 8192 + seg);
#pragma unroll
        for (int j = 0; j < 8; ++j) Vt[(seg + j) * 40 + row] = v[j];
      }
    }
    if (kt + 32 < 1024) { // prefetch next tiles toward L2/WGP$
      __builtin_prefetch(kbp + (size_t)(kt + 32 + (tid >> 3)) * 1024, 0, 1);
      __builtin_prefetch(vbp + (size_t)(kt + 32 + (tid >> 3)) * 8192, 0, 1);
    }
    wait_async0();
    __syncthreads();

    if (w == 0) {  // S^T = K . Q^T  (rows = keys, cols = queries) + softmax
      v8f st0 = {}, st1 = {};
#pragma unroll
      for (int kc = 0; kc < 4; ++kc) {
        const v16h bq = frag_b_trans(Qs + kc * 32, 136);
        st0 = wmma_f16(frag_a(Ks + kc * 32, 136), bq, st0);
        st1 = wmma_f16(frag_a(Ks + 16 * 136 + kc * 32, 136), bq, st1);
      }
      float tmax = -3.0e38f;
#pragma unroll
      for (int i = 0; i < 8; ++i) tmax = fmaxf(tmax, fmaxf(st0[i], st1[i]));
      tmax = fmaxf(tmax, __shfl_xor(tmax, 16, 32));
      const float newm = fmaxf(rowm, tmax);
      const float al = __expf(rowm - newm);
      float tsum = 0.0f;
      float p0[8], p1[8];
#pragma unroll
      for (int i = 0; i < 8; ++i) {
        p0[i] = __expf(st0[i] - newm);
        p1[i] = __expf(st1[i] - newm);
        tsum += p0[i] + p1[i];
      }
      tsum += __shfl_xor(tsum, 16, 32);
      lsum = lsum * al + tsum;
      rowm = newm;
#pragma unroll
      for (int i = 0; i < 8; ++i) {       // prob stored row-major [q][k]
        Ps[nl * 40 + 8 * half + i]      = (_Float16)p0[i];
        Ps[nl * 40 + 16 + 8 * half + i] = (_Float16)p1[i];
      }
      if (lane < 16) scaleS[lane] = al;
    }
    __syncthreads();

    float sc[8];
#pragma unroll
    for (int i = 0; i < 8; ++i) sc[i] = scaleS[8 * half + i];
    const v16h ap = frag_a(Ps, 40);
    const _Float16* Vt = Vs + w * (128 * 40);
#pragma unroll
    for (int c = 0; c < 8; ++c) {
      const v16h bvf = frag_b_trans(Vt + (c * 16) * 40, 40);
#pragma unroll
      for (int i = 0; i < 8; ++i) acc[c][i] *= sc[i];
      acc[c] = wmma_f16(ap, bvf, acc[c]);
    }
  }

  if (w == 0 && lane < 16) lsumS[lane] = lsum;
  __syncthreads();

  { // normalize and dump out[q][r][p] to LDS (reuses V staging space)
    float inv[8];
#pragma unroll
    for (int i = 0; i < 8; ++i) inv[i] = 1.0f / lsumS[8 * half + i];
#pragma unroll
    for (int c = 0; c < 8; ++c)
#pragma unroll
      for (int i = 0; i < 8; ++i) {
        const int q = 8 * half + i;
        out_lds[(q * 8 + w) * 128 + c * 16 + nl] = acc[c][i] * inv[i];
      }
  }
  for (int idx = tid; idx < 4096; idx += 256) WkvS[idx] = Wkv[idx];
  if (tid < 32) bkvS[tid] = bkv[tid];
  for (int idx = tid; idx < 512; idx += 256) {
    const int row = idx >> 5, j = idx & 31;
    qvS[idx] = (float)qvh[((size_t)b * 1024 + q0 + row) * 256 + h * 32 + j];
  }
  __syncthreads();

  if (tid < 128) {  // composition logits: dot(q_v, out@Wkv + bkv) per (q,r)
    const int q = tid >> 3, r = tid & 7;
    float kv[32];
#pragma unroll
    for (int j = 0; j < 32; ++j) kv[j] = bkvS[j];
    const float* orow = &out_lds[(q * 8 + r) * 128];
    for (int p = 0; p < 128; ++p) {
      const float v = orow[p];
#pragma unroll
      for (int j = 0; j < 32; ++j) kv[j] += v * WkvS[p * 32 + j];
    }
    float dot = 0.0f;
#pragma unroll
    for (int j = 0; j < 32; ++j) dot += qvS[q * 32 + j] * kv[j];
    compS[tid] = dot;
  }
  __syncthreads();
  if (tid < 16) {  // softmax over rules
    float m = -3.0e38f;
#pragma unroll
    for (int r = 0; r < 8; ++r) m = fmaxf(m, compS[tid * 8 + r]);
    float s = 0.0f, e[8];
#pragma unroll
    for (int r = 0; r < 8; ++r) { e[r] = __expf(compS[tid * 8 + r] - m); s += e[r]; }
    const float is = 1.0f / s;
#pragma unroll
    for (int r = 0; r < 8; ++r) wS[tid * 8 + r] = e[r] * is;
  }
  __syncthreads();
  { // weighted sum over rules -> out2 f16
    const int q = tid >> 4, ps = (tid & 15) * 8;
#pragma unroll
    for (int ii = 0; ii < 8; ++ii) {
      const int p = ps + ii;
      float s = 0.0f;
#pragma unroll
      for (int r = 0; r < 8; ++r) s += wS[q * 8 + r] * out_lds[(q * 8 + r) * 128 + p];
      out2h[((size_t)b * 1024 + q0 + q) * 1024 + h * 128 + p] = (_Float16)s;
    }
  }
}

// ---------------------------------------------------------------------------
extern "C" void kernel_launch(void* const* d_in, const int* in_sizes, int n_in,
                              void* d_out, int out_size, void* d_ws, size_t ws_size,
                              hipStream_t stream) {
  (void)in_sizes; (void)n_in; (void)out_size; (void)ws_size;
  const float* x   = (const float*)d_in[0];
  const float* Wq  = (const float*)d_in[1];
  const float* bq  = (const float*)d_in[2];
  const float* Wk  = (const float*)d_in[3];
  const float* bk  = (const float*)d_in[4];
  const float* Wv  = (const float*)d_in[5];
  const float* bv  = (const float*)d_in[6];
  const float* Wqv = (const float*)d_in[7];
  const float* bqv = (const float*)d_in[8];
  const float* Wkv = (const float*)d_in[9];
  const float* bkv = (const float*)d_in[10];
  const float* Wm  = (const float*)d_in[11];

  _Float16* base = (_Float16*)d_ws;
  size_t o = 0;
  _Float16* xh   = base + o; o += (size_t)8 * 1024 * 1024;
  _Float16* Wqh  = base + o; o += (size_t)1024 * 1024;
  _Float16* Wkh  = base + o; o += (size_t)1024 * 1024;
  _Float16* Wvh  = base + o; o += (size_t)1024 * 8192;
  _Float16* Wqvh = base + o; o += (size_t)1024 * 256;
  _Float16* Wmh  = base + o; o += (size_t)1024 * 1024;
  _Float16* qhb  = base + o; o += (size_t)8 * 1024 * 1024;
  _Float16* khb  = base + o; o += (size_t)8 * 1024 * 1024;
  _Float16* vhb  = base + o; o += (size_t)8 * 1024 * 8192;
  _Float16* qvhb = base + o; o += (size_t)8 * 1024 * 256;
  _Float16* o2h  = base + o; o += (size_t)8 * 1024 * 1024;

  auto cvt = [&](const float* src, _Float16* dst, size_t n) {
    cvt_f16<<<dim3((unsigned)((n + 255) / 256)), dim3(256), 0, stream>>>(src, dst, (int)n);
  };
  cvt(x,   xh,   (size_t)8 * 1024 * 1024);
  cvt(Wq,  Wqh,  (size_t)1024 * 1024);
  cvt(Wk,  Wkh,  (size_t)1024 * 1024);
  cvt(Wv,  Wvh,  (size_t)1024 * 8192);
  cvt(Wqv, Wqvh, (size_t)1024 * 256);
  cvt(Wm,  Wmh,  (size_t)1024 * 1024);

  const float sq  = 0.08838834764831845f;   // 1/sqrt(P=128)
  const float sqv = 0.17677669529663687f;   // 1/sqrt(QK=32)
  // q = (x@Wq + bq) * 1/sqrt(P)
  gemm16<false><<<dim3(8, 64), 256, 20480, stream>>>(xh, Wqh, bq, sq, qhb, nullptr, 8192, 1024, 1024);
  // k = x@Wk + bk
  gemm16<false><<<dim3(8, 64), 256, 20480, stream>>>(xh, Wkh, bk, 1.0f, khb, nullptr, 8192, 1024, 1024);
  // v = x@Wv + bv   (N = H*R*P = 8192)
  gemm16<false><<<dim3(64, 64), 256, 20480, stream>>>(xh, Wvh, bv, 1.0f, vhb, nullptr, 8192, 8192, 1024);
  // q_v = (x@Wqv + bqv) / sqrt(QK)
  gemm16<false><<<dim3(2, 64), 256, 20480, stream>>>(xh, Wqvh, bqv, sqv, qvhb, nullptr, 8192, 256, 1024);

  // fused attention + composition (dynamic LDS = 17664 + 81920 bytes)
  attn_fused<<<dim3(64, 64), 256, 99584, stream>>>(qhb, khb, vhb, qvhb, Wkv, bkv, o2h);

  // final merge: out = out2 @ Wm (f32 output)
  gemm16<true><<<dim3(8, 64), 256, 20480, stream>>>(o2h, Wmh, nullptr, 1.0f, nullptr,
                                                    (float*)d_out, 8192, 1024, 1024);
}
